// PicoDetHeadOLD_16088947491354
// MI455X (gfx1250) — matrix operand order repaired and checked by reading, not compiled, not run on hardware
//
#include <hip/hip_runtime.h>
#include <hip/hip_bf16.h>
#include <math.h>

typedef __attribute__((ext_vector_type(16))) _Float16 v16h;
typedef __attribute__((ext_vector_type(8)))  float    v8f;

#define NCLS 80
#define NTOT 5376   // 4096 + 1024 + 256 positions per image
#define BATCH 16
#define MAXDET 100

__device__ __forceinline__ float relu6f(float x) { return fminf(fmaxf(x, 0.0f), 6.0f); }
__device__ __forceinline__ float sigmoidf(float x) { return 1.0f / (1.0f + __expf(-x)); }

// B fragment (activations) for v_wmma_f32_16x16x32_f16:
// element e -> K = 16*half + e, N = lane&15 (posn = pos0 + n).
__device__ __forceinline__ void load_bfrags(const float* __restrict__ xb, int HW,
                                            int posn, int half, v16h bf[3]) {
#pragma unroll
  for (int s = 0; s < 3; ++s) {
#pragma unroll
    for (int e = 0; e < 16; ++e) {
      int ic = s * 32 + half * 16 + e;
      bf[s][e] = (_Float16)xb[(size_t)ic * HW + posn];
    }
  }
}

// A fragment (weights, row-major [rows,96]): M = lane&15 (row),
// VGPR v: K = 16*half + 2v+{0,1}.
__device__ __forceinline__ v16h load_afrag(const float* __restrict__ w, int row,
                                           int s, int half) {
  v16h a;
#pragma unroll
  for (int v = 0; v < 8; ++v) {
    int kb = 16 * half + 2 * v;
    const float* wr = w + (size_t)row * 96 + s * 32 + kb;
    a[2 * v]     = (_Float16)wr[0];
    a[2 * v + 1] = (_Float16)wr[1];
  }
  return a;
}

// ---------------------------------------------------------------------------
// Ghost pointwise: 1x1 conv 96->48 + BN + ReLU6 via WMMA.
// Writes channels [0,48) of out. grid.x = B*(HW/64), block = 128 (4 waves).
// ---------------------------------------------------------------------------
__global__ void ghost_pw_wmma(const float* __restrict__ x,  // [B,96,HW]
                              const float* __restrict__ w,  // [48,96]
                              const float* __restrict__ gm,
                              const float* __restrict__ bt,
                              const float* __restrict__ mn,
                              const float* __restrict__ vr,
                              float* __restrict__ out,      // [B,96,HW]
                              int HW) {
  const int lane = threadIdx.x & 31;
  const int wv   = threadIdx.x >> 5;
  const int tilesPerImg = HW >> 6;
  const int b    = blockIdx.x / tilesPerImg;
  const int pos0 = ((blockIdx.x % tilesPerImg) << 6) + wv * 16;
  const float* xb = x + (size_t)b * 96 * HW;
  float* ob = out + (size_t)b * 96 * HW;
  const int n    = lane & 15;
  const int half = lane >> 4;

  v16h bf[3];
  load_bfrags(xb, HW, pos0 + n, half, bf);

#pragma unroll
  for (int t = 0; t < 3; ++t) {           // oc tile (48 = 3 x 16)
    v8f acc = {};
#pragma unroll
    for (int s = 0; s < 3; ++s) {         // K step (96 = 3 x 32)
      v16h a = load_afrag(w, t * 16 + n, s, half);
      acc = __builtin_amdgcn_wmma_f32_16x16x32_f16(false, a, false, bf[s],
                                                   (short)0, acc, false, false);
    }
    // C layout: VGPR r -> oc = t*16 + r + 8*half, pos = lane&15. BN + ReLU6.
#pragma unroll
    for (int r = 0; r < 8; ++r) {
      int oc = t * 16 + r + 8 * half;
      float sc = gm[oc] * rsqrtf(vr[oc] + 1e-5f);
      float sh = bt[oc] - mn[oc] * sc;
      ob[(size_t)oc * HW + pos0 + n] = relu6f(acc[r] * sc + sh);
    }
  }
}

// ---------------------------------------------------------------------------
// Ghost cheap op: depthwise 3x3 (pad 1) on channels [0,48) + BN + ReLU6,
// writing channels [48,96) of the same buffer.
// ---------------------------------------------------------------------------
__global__ void ghost_dw(float* __restrict__ buf,          // [B,96,HW] in/out
                         const float* __restrict__ w2,     // [48,3,3]
                         const float* __restrict__ gm,
                         const float* __restrict__ bt,
                         const float* __restrict__ mn,
                         const float* __restrict__ vr,
                         int H, int W, int total) {
  int idx = blockIdx.x * blockDim.x + threadIdx.x;
  if (idx >= total) return;
  int HW = H * W;
  int p = idx % HW;
  int t = idx / HW;
  int c = t % 48;
  int b = t / 48;
  int h = p / W, wq = p % W;
  const float* src = buf + ((size_t)b * 96 + c) * HW;
  float acc = 0.0f;
#pragma unroll
  for (int dh = -1; dh <= 1; ++dh) {
    int hh = h + dh;
    if ((unsigned)hh >= (unsigned)H) continue;
#pragma unroll
    for (int dw = -1; dw <= 1; ++dw) {
      int ww = wq + dw;
      if ((unsigned)ww >= (unsigned)W) continue;
      acc += src[hh * W + ww] * w2[c * 9 + (dh + 1) * 3 + (dw + 1)];
    }
  }
  float sc = gm[c] * rsqrtf(vr[c] + 1e-5f);
  buf[((size_t)b * 96 + 48 + c) * HW + p] = relu6f(acc * sc + (bt[c] - mn[c] * sc));
}

// ---------------------------------------------------------------------------
// cls/obj head via WMMA: 80x96 GEMM -> per-position max/argmax (cross-half
// shfl merge, first-index tie-break), obj dot from the resident B fragments.
// grid.x = B*(HW/64), block = 128.
// ---------------------------------------------------------------------------
__global__ void cls_head_wmma(const float* __restrict__ feat,   // [B,96,HW]
                              const float* __restrict__ cls_w,  // [80,96]
                              const float* __restrict__ cls_b,  // [80]
                              const float* __restrict__ obj_w,  // [96]
                              const float* __restrict__ obj_b,  // [1]
                              float* __restrict__ conf,         // [B,NTOT]
                              int*   __restrict__ labels,       // [B,NTOT]
                              int HW, int base) {
  const int lane = threadIdx.x & 31;
  const int wv   = threadIdx.x >> 5;
  const int tilesPerImg = HW >> 6;
  const int b    = blockIdx.x / tilesPerImg;
  const int pos0 = ((blockIdx.x % tilesPerImg) << 6) + wv * 16;
  const float* xb = feat + (size_t)b * 96 * HW;
  const int n    = lane & 15;
  const int half = lane >> 4;

  v16h bf[3];
  load_bfrags(xb, HW, pos0 + n, half, bf);

  float best = -__builtin_inff();
  int arg = 0;
#pragma unroll
  for (int t = 0; t < 5; ++t) {           // 80 classes = 5 x 16
    v8f acc = {};
#pragma unroll
    for (int s = 0; s < 3; ++s) {
      v16h a = load_afrag(cls_w, t * 16 + n, s, half);
      acc = __builtin_amdgcn_wmma_f32_16x16x32_f16(false, a, false, bf[s],
                                                   (short)0, acc, false, false);
    }
#pragma unroll
    for (int r = 0; r < 8; ++r) {         // increasing oc within this lane
      int oc = t * 16 + r + 8 * half;
      float v = acc[r] + cls_b[oc];
      if (v > best) { best = v; arg = oc; }
    }
  }
  // merge the two lane-halves (first-index tie-break, matches jnp.argmax)
  float v2 = __shfl_xor(best, 16, 32);
  int   a2 = __shfl_xor(arg, 16, 32);
  if (v2 > best || (v2 == best && a2 < arg)) { best = v2; arg = a2; }

  // obj dot: each half owns ic in {s*32+16*half .. +15} — already in bf.
  float o = 0.0f;
#pragma unroll
  for (int s = 0; s < 3; ++s)
#pragma unroll
    for (int e = 0; e < 16; ++e)
      o += (float)bf[s][e] * obj_w[s * 32 + half * 16 + e];
  o += __shfl_xor(o, 16, 32);
  o += obj_b[0];

  if (half == 0) {
    int gp = base + pos0 + n;
    conf[(size_t)b * NTOT + gp]   = sigmoidf(best + o);
    labels[(size_t)b * NTOT + gp] = arg;
  }
}

// ---------------------------------------------------------------------------
// reg head via WMMA + fused DFL decode. C layout maps DFL sides directly:
// half0 holds l (t=0, oc 0..7) and r (t=1, oc 16..23);
// half1 holds t (oc 8..15) and b (oc 24..31); bin index = VGPR r.
// half0 lanes write x1/x2, half1 lanes write y1/y2 of the same box.
// grid.x = B*(HW/64), block = 128.
// ---------------------------------------------------------------------------
__global__ void reg_head_wmma(const float* __restrict__ feat,   // [B,96,HW]
                              const float* __restrict__ reg_w,  // [32,96]
                              const float* __restrict__ reg_b,  // [32]
                              float* __restrict__ boxes,        // [B,NTOT,4]
                              int H, int W, float stride, int base) {
  const int HW = H * W;
  const int lane = threadIdx.x & 31;
  const int wv   = threadIdx.x >> 5;
  const int tilesPerImg = HW >> 6;
  const int b    = blockIdx.x / tilesPerImg;
  const int pos0 = ((blockIdx.x % tilesPerImg) << 6) + wv * 16;
  const float* xb = feat + (size_t)b * 96 * HW;
  const int n    = lane & 15;
  const int half = lane >> 4;

  v16h bf[3];
  load_bfrags(xb, HW, pos0 + n, half, bf);

  float ex[2];
#pragma unroll
  for (int t = 0; t < 2; ++t) {           // 32 reg channels = 2 x 16
    v8f acc = {};
#pragma unroll
    for (int s = 0; s < 3; ++s) {
      v16h a = load_afrag(reg_w, t * 16 + n, s, half);
      acc = __builtin_amdgcn_wmma_f32_16x16x32_f16(false, a, false, bf[s],
                                                   (short)0, acc, false, false);
    }
    float v[8];
    float mx = -__builtin_inff();
#pragma unroll
    for (int r = 0; r < 8; ++r) {
      v[r] = acc[r] + reg_b[t * 16 + r + 8 * half];
      mx = fmaxf(mx, v[r]);
    }
    float se = 0.0f, sv = 0.0f;
#pragma unroll
    for (int r = 0; r < 8; ++r) {
      float e = __expf(v[r] - mx);
      se += e;
      sv += e * (float)r;
    }
    ex[t] = (sv / se) * stride;
  }

  int p = pos0 + n;
  int h = p / W, wq = p % W;
  float* o = boxes + ((size_t)b * NTOT + base + p) * 4;
  if (half == 0) {
    float cx = wq * stride + 0.5f * stride;
    o[0] = cx - ex[0];   // x1 = cx - l
    o[2] = cx + ex[1];   // x2 = cx + r
  } else {
    float cy = h * stride + 0.5f * stride;
    o[1] = cy - ex[0];   // y1 = cy - t
    o[3] = cy + ex[1];   // y2 = cy + b
  }
}

// ---------------------------------------------------------------------------
// Greedy NMS: one block per image, score vector + areas resident in LDS,
// block-wide argmax with first-index tie-break (matches jnp.argmax).
// ---------------------------------------------------------------------------
__global__ void nms_kernel(const float* __restrict__ boxes,  // [B,NTOT,4]
                           const float* __restrict__ conf,   // [B,NTOT]
                           const int*   __restrict__ labels, // [B,NTOT]
                           float* __restrict__ out) {
  __shared__ float c[NTOT];
  __shared__ float area[NTOT];
  __shared__ float redv[256];
  __shared__ int   redi[256];
  __shared__ float selbox[4];
  __shared__ int   seli;
  __shared__ float selc;

  int b = blockIdx.x;
  int tid = threadIdx.x;
  const float* bx = boxes + (size_t)b * NTOT * 4;
  const float* cf = conf + (size_t)b * NTOT;
  for (int i = tid; i < NTOT; i += 256) {
    float cc = cf[i];
    c[i] = (cc > 0.3f) ? cc : -__builtin_inff();
    area[i] = (bx[i * 4 + 2] - bx[i * 4 + 0]) * (bx[i * 4 + 3] - bx[i * 4 + 1]);
  }
  __syncthreads();

  float* ob = out + (size_t)b * MAXDET * 4;
  float* os = out + (size_t)BATCH * MAXDET * 4 + (size_t)b * MAXDET;
  float* ol = out + (size_t)BATCH * MAXDET * 4 + (size_t)BATCH * MAXDET + (size_t)b * MAXDET;

  for (int it = 0; it < MAXDET; ++it) {
    float bv = -__builtin_inff();
    int bi = 0x7fffffff;
    for (int i = tid; i < NTOT; i += 256) {
      float v = c[i];
      if (v > bv || (v == bv && i < bi)) { bv = v; bi = i; }
    }
    redv[tid] = bv;
    redi[tid] = bi;
    __syncthreads();
    for (int s = 128; s > 0; s >>= 1) {
      if (tid < s) {
        float v2 = redv[tid + s];
        int i2 = redi[tid + s];
        if (v2 > redv[tid] || (v2 == redv[tid] && i2 < redi[tid])) {
          redv[tid] = v2;
          redi[tid] = i2;
        }
      }
      __syncthreads();
    }
    if (tid == 0) {
      int i = redi[0];
      seli = i;
      selc = c[i];
      selbox[0] = bx[i * 4 + 0];
      selbox[1] = bx[i * 4 + 1];
      selbox[2] = bx[i * 4 + 2];
      selbox[3] = bx[i * 4 + 3];
    }
    __syncthreads();
    int i = seli;
    float ci = selc;
    float b0 = selbox[0], b1 = selbox[1], b2 = selbox[2], b3 = selbox[3];
    float ai = area[i];
    for (int p = tid; p < NTOT; p += 256) {
      float xx1 = fmaxf(b0, bx[p * 4 + 0]);
      float yy1 = fmaxf(b1, bx[p * 4 + 1]);
      float xx2 = fminf(b2, bx[p * 4 + 2]);
      float yy2 = fminf(b3, bx[p * 4 + 3]);
      float inter = fmaxf(xx2 - xx1, 0.0f) * fmaxf(yy2 - yy1, 0.0f);
      float iou = inter / (ai + area[p] - inter + 1e-9f);
      if (iou > 0.5f) c[p] = -__builtin_inff();
    }
    if (tid == 0) {
      c[i] = -__builtin_inff();
      bool keep = ci > 0.3f;
      ob[it * 4 + 0] = keep ? b0 : 0.0f;
      ob[it * 4 + 1] = keep ? b1 : 0.0f;
      ob[it * 4 + 2] = keep ? b2 : 0.0f;
      ob[it * 4 + 3] = keep ? b3 : 0.0f;
      os[it] = keep ? ci : 0.0f;
      ol[it] = keep ? (float)labels[(size_t)b * NTOT + i] : -1.0f;
    }
    __syncthreads();
  }
}

// ---------------------------------------------------------------------------
// Host side
// ---------------------------------------------------------------------------
struct GhostP {
  const float *bt1, *gm1, *mn1, *vr1;  // bn1: beta,gamma,mean,var (alpha order)
  const float *bt2, *gm2, *mn2, *vr2;  // bn2
  const float *w1, *w2;
};

static GhostP ghost_at(void* const* d_in, int i0) {
  GhostP g;
  g.bt1 = (const float*)d_in[i0 + 0];
  g.gm1 = (const float*)d_in[i0 + 1];
  g.mn1 = (const float*)d_in[i0 + 2];
  g.vr1 = (const float*)d_in[i0 + 3];
  g.bt2 = (const float*)d_in[i0 + 4];
  g.gm2 = (const float*)d_in[i0 + 5];
  g.mn2 = (const float*)d_in[i0 + 6];
  g.vr2 = (const float*)d_in[i0 + 7];
  g.w1  = (const float*)d_in[i0 + 8];
  g.w2  = (const float*)d_in[i0 + 9];
  return g;
}

static void run_ghost(const float* in, float* outbuf, const GhostP& g,
                      int H, int W, hipStream_t stream) {
  int HW = H * W;
  dim3 gpw(BATCH * (HW / 64));
  ghost_pw_wmma<<<gpw, 128, 0, stream>>>(in, g.w1, g.gm1, g.bt1, g.mn1, g.vr1, outbuf, HW);
  int total = BATCH * 48 * HW;
  dim3 gdw((total + 255) / 256);
  ghost_dw<<<gdw, 256, 0, stream>>>(outbuf, g.w2, g.gm2, g.bt2, g.mn2, g.vr2, H, W, total);
}

extern "C" void kernel_launch(void* const* d_in, const int* in_sizes, int n_in,
                              void* d_out, int out_size, void* d_ws, size_t ws_size,
                              hipStream_t stream) {
  (void)in_sizes; (void)n_in; (void)out_size; (void)ws_size;

  const float* feat[3] = {(const float*)d_in[0], (const float*)d_in[1], (const float*)d_in[2]};
  // pytree (alphabetical) flatten: cls_conv[0](3..12), cls_conv[1](13..22),
  // heads[0..2](23..40, 6 leaves each: cls_b, cls_w, obj_b, obj_w, reg_b, reg_w),
  // reg_conv[0](41..50), reg_conv[1](51..60)
  GhostP clsG0 = ghost_at(d_in, 3);
  GhostP clsG1 = ghost_at(d_in, 13);
  GhostP regG0 = ghost_at(d_in, 41);
  GhostP regG1 = ghost_at(d_in, 51);

  const int headBase = 23;
  const int Hs[3] = {64, 32, 16};
  const float strides[3] = {8.0f, 16.0f, 32.0f};
  const int bases[3] = {0, 4096, 5120};

  // Workspace carve-up (floats)
  float* ws = (float*)d_ws;
  const size_t bufElems = (size_t)BATCH * 96 * 4096;  // largest level
  float* bufA = ws;
  float* bufB = bufA + bufElems;
  float* conf = bufB + bufElems;
  int* labels = (int*)(conf + (size_t)BATCH * NTOT);
  float* boxes = (float*)(labels + (size_t)BATCH * NTOT);

  for (int lv = 0; lv < 3; ++lv) {
    int H = Hs[lv], W = Hs[lv], HW = H * W;
    const float* cls_b = (const float*)d_in[headBase + lv * 6 + 0];
    const float* cls_w = (const float*)d_in[headBase + lv * 6 + 1];
    const float* obj_b = (const float*)d_in[headBase + lv * 6 + 2];
    const float* obj_w = (const float*)d_in[headBase + lv * 6 + 3];
    const float* reg_b = (const float*)d_in[headBase + lv * 6 + 4];
    const float* reg_w = (const float*)d_in[headBase + lv * 6 + 5];

    // cls branch: ghost(ghost(f)) -> conf/labels
    run_ghost(feat[lv], bufA, clsG0, H, W, stream);
    run_ghost(bufA, bufB, clsG1, H, W, stream);
    cls_head_wmma<<<dim3(BATCH * (HW / 64)), 128, 0, stream>>>(
        bufB, cls_w, cls_b, obj_w, obj_b, conf, labels, HW, bases[lv]);

    // reg branch: ghost(ghost(f)) -> boxes (DFL decode fused)
    run_ghost(feat[lv], bufA, regG0, H, W, stream);
    run_ghost(bufA, bufB, regG1, H, W, stream);
    reg_head_wmma<<<dim3(BATCH * (HW / 64)), 128, 0, stream>>>(
        bufB, reg_w, reg_b, boxes, H, W, strides[lv], bases[lv]);
  }

  nms_kernel<<<dim3(BATCH), 256, 0, stream>>>(boxes, conf, labels, (float*)d_out);
}
